// Struct2Seq_37383395344843
// MI455X (gfx1250) — matrix operand order repaired and checked by reading, not compiled, not run on hardware
//
#include <hip/hip_runtime.h>
#include <cstdint>
#include <cstddef>

typedef __bf16 bf16;
typedef __attribute__((ext_vector_type(16))) __bf16 v16bf;
typedef __attribute__((ext_vector_type(8)))  __bf16 v8bf;
typedef __attribute__((ext_vector_type(8)))  float  v8f;

#define NEG_INF_F (-3.402823466e38f)

// ---------------------------------------------------------------------------
// WMMA fragment helpers (CDNA5 wave32 layouts, ISA 7.12.2)
//  A 16x32 bf16: lane l=lane&15 -> row l, h=lane>>4 selects K-halves:
//    elems 0..7  -> K = k0 + 8h + e ; elems 8..15 -> K = k0 + 16 + 8h + (e-8)
//  B 32x16 bf16 (prepacked B^T, row-major [N][K]): lane -> column, same K
//    split as A: elems 0..7 -> K = k0 + 8h + e ; elems 8..15 -> K = k0 + 16 + 8h + (e-8)
//  C/D f32: elem r -> row (8h + r), col (lane&15)
// ---------------------------------------------------------------------------
__device__ __forceinline__ v16bf zero16() {
  v16bf a;
#pragma unroll
  for (int i = 0; i < 16; ++i) a[i] = (bf16)0.0f;
  return a;
}

__device__ __forceinline__ v16bf load_afrag(const bf16* __restrict__ row, int k0, int h) {
  v8bf lo = *(const v8bf*)(row + k0 + 8 * h);
  v8bf hi = *(const v8bf*)(row + k0 + 16 + 8 * h);
  v16bf a;
#pragma unroll
  for (int i = 0; i < 8; ++i) { a[i] = lo[i]; a[i + 8] = hi[i]; }
  return a;
}

__device__ __forceinline__ v16bf load_afrag(const float* __restrict__ row, int k0, int h) {
  v8f lo = *(const v8f*)(row + k0 + 8 * h);
  v8f hi = *(const v8f*)(row + k0 + 16 + 8 * h);
  v16bf a;
#pragma unroll
  for (int i = 0; i < 8; ++i) { a[i] = (bf16)lo[i]; a[i + 8] = (bf16)hi[i]; }
  return a;
}

__device__ __forceinline__ v16bf load_bfrag(const bf16* __restrict__ colp, int k0, int h) {
  v8bf lo = *(const v8bf*)(colp + k0 + 8 * h);
  v8bf hi = *(const v8bf*)(colp + k0 + 16 + 8 * h);
  v16bf b;
#pragma unroll
  for (int i = 0; i < 8; ++i) { b[i] = lo[i]; b[i + 8] = hi[i]; }
  return b;
}

// ---------------------------------------------------------------------------
// A-stationary WMMA GEMM: one wave -> 16 rows x (16*NT) cols.
// A-frag loaded once per k-step, reused for NT WMMAs (cuts A HBM traffic NTx).
// ---------------------------------------------------------------------------
template <typename AT, int NT>
__global__ __launch_bounds__(128) void gemm_kernel(
    const AT* __restrict__ A, const bf16* __restrict__ Bt,
    const float* __restrict__ bias, const float* __restrict__ resid,
    float* __restrict__ outF, bf16* __restrict__ outB,
    int M, int N, int K, int relu) {
  const int wave = threadIdx.x >> 5;
  const int lane = threadIdx.x & 31;
  const int l = lane & 15, h = lane >> 4;
  const int chunks = (N + 16 * NT - 1) / (16 * NT);
  const long tilesM = (long)(M + 15) >> 4;
  const long t = (long)blockIdx.x * 4 + wave;
  if (t >= tilesM * chunks) return;  // whole-wave exit, EXEC stays full for WMMA
  const int tm = (int)(t / chunks);
  const int cbase = (int)(t % chunks) * (16 * NT);

  const AT*   arow = A  + (size_t)(tm * 16 + l) * K;
  const bf16* bcol = Bt + (size_t)(cbase + l) * K;  // +j*16*K per col tile

  v8f acc[NT];
#pragma unroll
  for (int j = 0; j < NT; ++j)
#pragma unroll
    for (int i = 0; i < 8; ++i) acc[j][i] = 0.0f;

  for (int k0 = 0; k0 < K; k0 += 32) {
    if (k0 + 64 < K) __builtin_prefetch(arow + k0 + 64);  // global_prefetch_b8
    const v16bf a = load_afrag(arow, k0, h);
#pragma unroll
    for (int j = 0; j < NT; ++j) {
      const v16bf b = load_bfrag(bcol + (size_t)j * 16 * K, k0, h);
      acc[j] = __builtin_amdgcn_wmma_f32_16x16x32_bf16(false, a, false, b, (short)0,
                                                       acc[j], false, false);
    }
  }

#pragma unroll
  for (int j = 0; j < NT; ++j) {
    const int col = cbase + j * 16 + l;
    const bool cok = col < N;
    const float bvv = (bias && cok) ? bias[col] : 0.0f;
#pragma unroll
    for (int r = 0; r < 8; ++r) {
      const int row = tm * 16 + h * 8 + r;
      if (cok && row < M) {
        float v = acc[j][r] + bvv;
        if (resid) v += resid[(size_t)row * N + col];
        if (relu) v = fmaxf(v, 0.0f);
        if (outF) outF[(size_t)row * N + col] = v;
        if (outB) outB[(size_t)row * N + col] = (bf16)v;
      }
    }
  }
}

// ---------------------------------------------------------------------------
// Fused gather-concat K+V projection (one pass over h_E produces both).
// Encoder (DEC=0, KD=256): row(e) = [ h_E[e] | h_V[gather] ]
// Decoder (DEC=1, KD=384): row(e) = mask(n)*( ar ? [h_E|h_S_g|h_V_g]
//                                               : [h_E|  0  |h_Venc_g] )
// Never materializes h_EV/h_ESV (saves ~300MB of HBM traffic per layer).
// One wave: 16 edge-rows x 128 cols x {K,V} = 16 WMMA accum tiles.
// ---------------------------------------------------------------------------
template <int KD, bool DEC>
__global__ __launch_bounds__(128) void kvproj_kernel(
    const bf16* __restrict__ hE, const bf16* __restrict__ g1,
    const bf16* __restrict__ g2a, const bf16* __restrict__ g2b,
    const int* __restrict__ E_idx, const float* __restrict__ mask,
    const bf16* __restrict__ BtK, const bf16* __restrict__ BtV,
    bf16* __restrict__ outK, bf16* __restrict__ outV,
    int M, int Kn, int Nnodes) {
  const int wave = threadIdx.x >> 5;
  const int lane = threadIdx.x & 31;
  const int l = lane & 15, h = lane >> 4;
  const long tm = (long)blockIdx.x * 4 + wave;
  if (tm >= (long)(M >> 4)) return;

  const int e  = (int)tm * 16 + l;  // edge row owned by this lane
  const int ng = e / Kn;            // global node row (b*N + n)
  const int b  = ng / Nnodes;
  const int n  = ng - b * Nnodes;
  const int j  = E_idx[e];
  const size_t jg = (size_t)b * Nnodes + j;

  const bf16* p0 = hE + (size_t)e * 128;
  const bf16* p1;
  const bf16* p2 = nullptr;
  float mv = 1.0f;
  if (!DEC) {
    p1 = g1 + jg * 128;                    // gathered h_V
  } else {
    const int ar = (j < n) ? 1 : 0;        // neighbor strictly before n
    mv = mask[ng];
    p1 = ar ? (g1 + jg * 128) : nullptr;   // h_S or zeros
    p2 = (ar ? g2a : g2b) + jg * 128;      // h_V current or frozen encoder h_V
  }

  const bf16* bK = BtK + (size_t)l * KD;
  const bf16* bV = BtV + (size_t)l * KD;

  v8f accK[8], accV[8];
#pragma unroll
  for (int t = 0; t < 8; ++t)
#pragma unroll
    for (int i = 0; i < 8; ++i) { accK[t][i] = 0.0f; accV[t][i] = 0.0f; }

#pragma unroll
  for (int part = 0; part < KD / 128; ++part) {
    const bf16* p = (part == 0) ? p0 : ((part == 1) ? p1 : p2);
#pragma unroll
    for (int ko = 0; ko < 128; ko += 32) {
      v16bf a;
      if (p) a = load_afrag(p, ko, h); else a = zero16();
      if (DEC) {
#pragma unroll
        for (int i = 0; i < 16; ++i) a[i] = (bf16)((float)a[i] * mv);
      }
      const int k0 = part * 128 + ko;
#pragma unroll
      for (int t = 0; t < 8; ++t) {
        const v16bf bb = load_bfrag(bK + (size_t)t * 16 * KD, k0, h);
        accK[t] = __builtin_amdgcn_wmma_f32_16x16x32_bf16(false, a, false, bb,
                                                          (short)0, accK[t], false, false);
      }
#pragma unroll
      for (int t = 0; t < 8; ++t) {
        const v16bf bb = load_bfrag(bV + (size_t)t * 16 * KD, k0, h);
        accV[t] = __builtin_amdgcn_wmma_f32_16x16x32_bf16(false, a, false, bb,
                                                          (short)0, accV[t], false, false);
      }
    }
  }

#pragma unroll
  for (int t = 0; t < 8; ++t) {
    const int col = t * 16 + l;
#pragma unroll
    for (int r = 0; r < 8; ++r) {
      const size_t row = (size_t)tm * 16 + h * 8 + r;
      outK[row * 128 + col] = (bf16)accK[t][r];
      outV[row * 128 + col] = (bf16)accV[t][r];
    }
  }
}

// ---------------------------------------------------------------------------
// Neighbor attention, one wave per node (lane = neighbor slot, K=30 <= 32).
// K/V rows for the node are staged into LDS with gfx1250 async-to-LDS b128
// copies (ASYNCcnt), then consumed wave-synchronously after s_wait_asynccnt.
// K rows padded to 272B so per-lane row walks spread across LDS banks;
// V phase-2 access (8B/lane, consecutive) is conflict-free unpadded.
// ---------------------------------------------------------------------------
#define KROWPAD 272
#define SK_BYTES (30 * KROWPAD)  // 8160
#define SV_BYTES (30 * 256)      // 7680
#define SWV_BYTES (SK_BYTES + SV_BYTES)

__global__ __launch_bounds__(128) void attn_kernel(
    const float* __restrict__ Qf, const bf16* __restrict__ Kt,
    const bf16* __restrict__ Vt, const int* __restrict__ E_idx,
    const float* __restrict__ mask, bf16* __restrict__ upd,
    int NN, int Kn, int Nnodes, int useMask) {
  __shared__ __align__(16) char smem[4 * SWV_BYTES];
  const int wave = threadIdx.x >> 5;
  const int lane = threadIdx.x & 31;
  const int g = blockIdx.x * 4 + wave;
  if (g >= NN) return;

  char* sK = smem + wave * SWV_BYTES;
  char* sV = sK + SK_BYTES;
  const unsigned sKoff = (unsigned)(uintptr_t)sK;  // LDS byte offset (low 32 bits)
  const unsigned sVoff = (unsigned)(uintptr_t)sV;
  const char* gK = (const char*)(Kt + (size_t)g * Kn * 128);
  const char* gV = (const char*)(Vt + (size_t)g * Kn * 128);

  // Stage 30 rows x 256B of K (padded) and V (dense): 480 chunks each,
  // 15 rounds of one b128 async copy per lane per matrix.
#pragma unroll
  for (int i = 0; i < 15; ++i) {
    const int c = lane + 32 * i;
    const int r = c >> 4, cc = c & 15;
    const unsigned ldsK = sKoff + r * KROWPAD + cc * 16;
    const unsigned long long gaK = (unsigned long long)(uintptr_t)(gK + c * 16);
    asm volatile("global_load_async_to_lds_b128 %0, %1, off"
                 :: "v"(ldsK), "v"(gaK) : "memory");
    const unsigned ldsV = sVoff + c * 16;
    const unsigned long long gaV = (unsigned long long)(uintptr_t)(gV + c * 16);
    asm volatile("global_load_async_to_lds_b128 %0, %1, off"
                 :: "v"(ldsV), "v"(gaV) : "memory");
  }

  // Per-lane neighbor metadata while copies are in flight.
  const int k = lane;
  const bool valid = k < Kn;
  const size_t e = (size_t)g * Kn + (valid ? k : 0);
  float ma = 1.0f;
  if (useMask) {
    const int b = g / Nnodes;
    ma = mask[g] * mask[(size_t)b * Nnodes + E_idx[e]];
  }
  const bool act = valid && (!useMask || ma > 0.0f);

  asm volatile("s_wait_asynccnt 0x0" ::: "memory");

  // Phase 1: logits for this lane's neighbor, all 4 heads, K row from LDS.
  const float* qp = Qf + (size_t)g * 128;
  const bf16* kr = (const bf16*)(sK + k * KROWPAD);
  float lgs[4] = {0.f, 0.f, 0.f, 0.f};
#pragma unroll
  for (int c = 0; c < 8; ++c) {        // 16 values per chunk, head = c>>1
    const v8bf k0 = *(const v8bf*)(kr + c * 16);
    const v8bf k1 = *(const v8bf*)(kr + c * 16 + 8);
    const int hh = c >> 1;
    float s = 0.f;
#pragma unroll
    for (int tt = 0; tt < 8; ++tt) {
      s += qp[c * 16 + tt]     * (float)k0[tt];
      s += qp[c * 16 + 8 + tt] * (float)k1[tt];
    }
    lgs[hh] += s;
  }
  const float scale = 0.17677669529663689f;  // 1/sqrt(32)

  float at[4];
#pragma unroll
  for (int hh = 0; hh < 4; ++hh) {
    float x = act ? lgs[hh] * scale : NEG_INF_F;
    float mx = x;
#pragma unroll
    for (int m = 16; m >= 1; m >>= 1) mx = fmaxf(mx, __shfl_xor(mx, m));
    float ex = act ? __expf(x - mx) : 0.0f;
    float s = ex;
#pragma unroll
    for (int m = 16; m >= 1; m >>= 1) s += __shfl_xor(s, m);
    float a = ex / fmaxf(s, 1e-20f);
    if (useMask) a *= ma;  // attend = softmax(masked logits) * m
    at[hh] = a;
  }

  // Phase 2: lane owns dims d = lane*4 .. lane*4+3 (head = lane>>3); V from LDS.
  const int hsel = lane >> 3;
  float o0 = 0.f, o1 = 0.f, o2 = 0.f, o3 = 0.f;
  for (int kk = 0; kk < Kn; ++kk) {
    const float w0 = __shfl(at[0], kk);
    const float w1 = __shfl(at[1], kk);
    const float w2 = __shfl(at[2], kk);
    const float w3 = __shfl(at[3], kk);
    const float w = (hsel == 0) ? w0 : (hsel == 1) ? w1 : (hsel == 2) ? w2 : w3;
    const bf16* vp = (const bf16*)(sV + kk * 256) + lane * 4;
    o0 += w * (float)vp[0];
    o1 += w * (float)vp[1];
    o2 += w * (float)vp[2];
    o3 += w * (float)vp[3];
  }
  bf16* op = upd + (size_t)g * 128 + lane * 4;
  op[0] = (bf16)o0; op[1] = (bf16)o1; op[2] = (bf16)o2; op[3] = (bf16)o3;
}

// ---------------------------------------------------------------------------
// LayerNorm over last dim (128): one wave per row, shfl reductions.
// ---------------------------------------------------------------------------
__global__ __launch_bounds__(128) void ln_kernel(
    const float* __restrict__ x, const float* __restrict__ gw,
    const float* __restrict__ bw, const float* __restrict__ mask,
    float* __restrict__ outF, bf16* __restrict__ outB, int M) {
  const int wave = threadIdx.x >> 5;
  const int lane = threadIdx.x & 31;
  const int row = blockIdx.x * 4 + wave;
  if (row >= M) return;
  const float* xr = x + (size_t)row * 128;
  float v[4];
  float s = 0.f;
#pragma unroll
  for (int i = 0; i < 4; ++i) { v[i] = xr[lane * 4 + i]; s += v[i]; }
#pragma unroll
  for (int m = 16; m >= 1; m >>= 1) s += __shfl_xor(s, m);
  const float mu = s * (1.0f / 128.0f);
  float q = 0.f;
#pragma unroll
  for (int i = 0; i < 4; ++i) { v[i] -= mu; q += v[i] * v[i]; }
#pragma unroll
  for (int m = 16; m >= 1; m >>= 1) q += __shfl_xor(q, m);
  const float inv = rsqrtf(q * (1.0f / 128.0f) + 1e-6f);
  const float mv = mask ? mask[row] : 1.0f;
#pragma unroll
  for (int i = 0; i < 4; ++i) {
    const int d = lane * 4 + i;
    float y = (gw[d] * v[i] * inv + bw[d]) * mv;
    outF[(size_t)row * 128 + d] = y;
    outB[(size_t)row * 128 + d] = (bf16)y;
  }
}

// Weight prep: f32 [K,N] -> bf16 B^T [Npad,K] (zero pad cols >= N)
__global__ void wprep_kernel(const float* __restrict__ src, bf16* __restrict__ dst,
                             int K, int N, int Npad) {
  const int i = blockIdx.x * blockDim.x + threadIdx.x;
  if (i >= Npad * K) return;
  const int n = i / K, k = i - n * K;
  dst[i] = (n < N) ? (bf16)src[(size_t)k * N + n] : (bf16)0.0f;
}

// Sequence embedding lookup: h_S = Ws[S]
__global__ void embed_kernel(const int* __restrict__ S, const float* __restrict__ Ws,
                             bf16* __restrict__ hS, int M) {
  const int i = blockIdx.x * blockDim.x + threadIdx.x;
  if (i >= M * 128) return;
  const int r = i >> 7, d = i & 127;
  hS[i] = (bf16)Ws[(size_t)S[r] * 128 + d];
}

// log_softmax over 20 classes, one thread per row
__global__ void lsm_kernel(const float* __restrict__ logits, float* __restrict__ out,
                           int M, int C) {
  const int r = blockIdx.x * blockDim.x + threadIdx.x;
  if (r >= M) return;
  const float* lr = logits + (size_t)r * C;
  float mx = NEG_INF_F;
  for (int c = 0; c < C; ++c) mx = fmaxf(mx, lr[c]);
  float s = 0.f;
  for (int c = 0; c < C; ++c) s += __expf(lr[c] - mx);
  const float lse = mx + __logf(s);
  for (int c = 0; c < C; ++c) out[(size_t)r * C + c] = lr[c] - lse;
}

// ---------------------------------------------------------------------------
extern "C" void kernel_launch(void* const* d_in, const int* in_sizes, int n_in,
                              void* d_out, int out_size, void* d_ws, size_t ws_size,
                              hipStream_t stream) {
  (void)in_sizes; (void)n_in; (void)out_size; (void)ws_size;
  constexpr int B = 8, N = 1024, Kn = 30, H = 128, FF = 512, L = 3, VOC = 20;
  const int NN = B * N;    // 8192 node rows
  const int ME = NN * Kn;  // 245760 edge rows

  const float* V      = (const float*)d_in[0];
  const float* E      = (const float*)d_in[1];
  const int*   E_idx  = (const int*)d_in[2];
  const int*   S      = (const int*)d_in[3];
  const float* mask   = (const float*)d_in[4];
  const float* Wv     = (const float*)d_in[5];
  const float* bv     = (const float*)d_in[6];
  const float* We     = (const float*)d_in[7];
  const float* be     = (const float*)d_in[8];
  const float* Ws     = (const float*)d_in[9];
  const float* eWQ    = (const float*)d_in[10];
  const float* eWK    = (const float*)d_in[11];
  const float* eWV    = (const float*)d_in[12];
  const float* eWO    = (const float*)d_in[13];
  const float* eWin   = (const float*)d_in[14];
  const float* ebin   = (const float*)d_in[15];
  const float* eWout  = (const float*)d_in[16];
  const float* ebout  = (const float*)d_in[17];
  const float* eg     = (const float*)d_in[18];
  const float* eb     = (const float*)d_in[19];
  const float* dWQ    = (const float*)d_in[20];
  const float* dWK    = (const float*)d_in[21];
  const float* dWV    = (const float*)d_in[22];
  const float* dWO    = (const float*)d_in[23];
  const float* dWin   = (const float*)d_in[24];
  const float* dbin   = (const float*)d_in[25];
  const float* dWout  = (const float*)d_in[26];
  const float* dbout  = (const float*)d_in[27];
  const float* dg     = (const float*)d_in[28];
  const float* db     = (const float*)d_in[29];
  const float* W_out  = (const float*)d_in[30];
  const float* b_out  = (const float*)d_in[31];

  // ---- workspace carve (deterministic bump allocator) ----
  char* base = (char*)d_ws;
  size_t off = 0;
  auto alloc = [&](size_t bytes) -> void* {
    off = (off + 255) & ~(size_t)255;
    void* r = base + off;
    off += bytes;
    return r;
  };
  bf16*  hE    = (bf16*)alloc((size_t)ME * H * 2);
  bf16*  Kt    = (bf16*)alloc((size_t)ME * H * 2);
  bf16*  Vt    = (bf16*)alloc((size_t)ME * H * 2);
  float* hVf   = (float*)alloc((size_t)NN * H * 4);
  float* tmpF  = (float*)alloc((size_t)NN * H * 4);
  float* Qf    = (float*)alloc((size_t)NN * H * 4);
  bf16*  hVb   = (bf16*)alloc((size_t)NN * H * 2);
  bf16*  hVenc = (bf16*)alloc((size_t)NN * H * 2);
  bf16*  hS    = (bf16*)alloc((size_t)NN * H * 2);
  bf16*  updB  = (bf16*)alloc((size_t)NN * H * 2);
  bf16*  midB  = (bf16*)alloc((size_t)NN * FF * 2);
  float* logF  = (float*)alloc((size_t)NN * VOC * 4);
  bf16* WvT = (bf16*)alloc((size_t)H * H * 2);
  bf16* WeT = (bf16*)alloc((size_t)H * H * 2);
  bf16 *eQT[L], *eKT[L], *eVT[L], *eOT[L], *eInT[L], *eOutT[L];
  bf16 *dQT[L], *dKT[L], *dVT[L], *dOT[L], *dInT[L], *dOutT[L];
  for (int l = 0; l < L; ++l) {
    eQT[l]   = (bf16*)alloc((size_t)H * H * 2);
    eKT[l]   = (bf16*)alloc((size_t)H * 2 * H * 2);
    eVT[l]   = (bf16*)alloc((size_t)H * 2 * H * 2);
    eOT[l]   = (bf16*)alloc((size_t)H * H * 2);
    eInT[l]  = (bf16*)alloc((size_t)FF * H * 2);
    eOutT[l] = (bf16*)alloc((size_t)H * FF * 2);
  }
  for (int l = 0; l < L; ++l) {
    dQT[l]   = (bf16*)alloc((size_t)H * H * 2);
    dKT[l]   = (bf16*)alloc((size_t)H * 3 * H * 2);
    dVT[l]   = (bf16*)alloc((size_t)H * 3 * H * 2);
    dOT[l]   = (bf16*)alloc((size_t)H * H * 2);
    dInT[l]  = (bf16*)alloc((size_t)FF * H * 2);
    dOutT[l] = (bf16*)alloc((size_t)H * FF * 2);
  }
  bf16* WoT = (bf16*)alloc((size_t)32 * H * 2);

  // ---- weight prep ----
  auto wprep = [&](const float* src, bf16* dst, int K, int Ncols, int Npad) {
    const int total = Npad * K;
    wprep_kernel<<<(total + 255) / 256, 256, 0, stream>>>(src, dst, K, Ncols, Npad);
  };
  wprep(Wv, WvT, H, H, H);
  wprep(We, WeT, H, H, H);
  for (int l = 0; l < L; ++l) {
    wprep(eWQ   + (size_t)l * H * H,     eQT[l],   H,     H,  H);
    wprep(eWK   + (size_t)l * 2 * H * H, eKT[l],   2 * H, H,  H);
    wprep(eWV   + (size_t)l * 2 * H * H, eVT[l],   2 * H, H,  H);
    wprep(eWO   + (size_t)l * H * H,     eOT[l],   H,     H,  H);
    wprep(eWin  + (size_t)l * H * FF,    eInT[l],  H,     FF, FF);
    wprep(eWout + (size_t)l * FF * H,    eOutT[l], FF,    H,  H);
    wprep(dWQ   + (size_t)l * H * H,     dQT[l],   H,     H,  H);
    wprep(dWK   + (size_t)l * 3 * H * H, dKT[l],   3 * H, H,  H);
    wprep(dWV   + (size_t)l * 3 * H * H, dVT[l],   3 * H, H,  H);
    wprep(dWO   + (size_t)l * H * H,     dOT[l],   H,     H,  H);
    wprep(dWin  + (size_t)l * H * FF,    dInT[l],  H,     FF, FF);
    wprep(dWout + (size_t)l * FF * H,    dOutT[l], FF,    H,  H);
  }
  wprep(W_out, WoT, H, VOC, 32);

  // ---- launchers ----
  auto gemmf = [&](const float* A, const bf16* Bt, const float* bias,
                   const float* resid, float* oF, bf16* oB, int M, int Nc, int K,
                   int relu) {
    const long waves = (long)((M + 15) / 16) * ((Nc + 127) / 128);
    gemm_kernel<float, 8><<<dim3((unsigned)((waves + 3) / 4)), 128, 0, stream>>>(
        A, Bt, bias, resid, oF, oB, M, Nc, K, relu);
  };
  auto gemmb = [&](const bf16* A, const bf16* Bt, const float* bias,
                   const float* resid, float* oF, bf16* oB, int M, int Nc, int K,
                   int relu) {
    const long waves = (long)((M + 15) / 16) * ((Nc + 127) / 128);
    gemm_kernel<bf16, 8><<<dim3((unsigned)((waves + 3) / 4)), 128, 0, stream>>>(
        A, Bt, bias, resid, oF, oB, M, Nc, K, relu);
  };
  const unsigned kvblocks = (unsigned)((ME / 16 + 3) / 4);

  // ---- input projections ----
  gemmf(V, WvT, bv, nullptr, hVf, hVb, NN, H, H, 0);     // h_V (f32 + bf16)
  gemmf(E, WeT, be, nullptr, nullptr, hE, ME, H, H, 0);  // h_E (bf16)

  // ---- encoder ----
  for (int l = 0; l < L; ++l) {
    kvproj_kernel<256, false><<<kvblocks, 128, 0, stream>>>(
        hE, hVb, nullptr, nullptr, E_idx, nullptr, eKT[l], eVT[l], Kt, Vt, ME, Kn, N);
    gemmb(hVb, eQT[l], nullptr, nullptr, Qf, nullptr, NN, H, H, 0);
    attn_kernel<<<NN / 4, 128, 0, stream>>>(Qf, Kt, Vt, E_idx, mask, updB, NN, Kn, N, 1);
    gemmb(updB, eOT[l], nullptr, hVf, tmpF, nullptr, NN, H, H, 0);
    ln_kernel<<<NN / 4, 128, 0, stream>>>(tmpF, eg + ((size_t)l * 2 + 0) * H,
                                          eb + ((size_t)l * 2 + 0) * H, nullptr,
                                          hVf, hVb, NN);
    gemmb(hVb, eInT[l], ebin + (size_t)l * FF, nullptr, nullptr, midB, NN, FF, H, 1);
    gemmb(midB, eOutT[l], ebout + (size_t)l * H, hVf, tmpF, nullptr, NN, H, FF, 0);
    ln_kernel<<<NN / 4, 128, 0, stream>>>(tmpF, eg + ((size_t)l * 2 + 1) * H,
                                          eb + ((size_t)l * 2 + 1) * H, mask,
                                          hVf, hVb, NN);
  }

  // freeze encoder h_V for autoregressive forward features; embed sequence
  hipMemcpyAsync(hVenc, hVb, (size_t)NN * H * 2, hipMemcpyDeviceToDevice, stream);
  embed_kernel<<<(NN * H + 255) / 256, 256, 0, stream>>>(S, Ws, hS, NN);

  // ---- decoder ----
  for (int l = 0; l < L; ++l) {
    kvproj_kernel<384, true><<<kvblocks, 128, 0, stream>>>(
        hE, hS, hVb, hVenc, E_idx, mask, dKT[l], dVT[l], Kt, Vt, ME, Kn, N);
    gemmb(hVb, dQT[l], nullptr, nullptr, Qf, nullptr, NN, H, H, 0);
    attn_kernel<<<NN / 4, 128, 0, stream>>>(Qf, Kt, Vt, E_idx, nullptr, updB, NN, Kn, N, 0);
    gemmb(updB, dOT[l], nullptr, hVf, tmpF, nullptr, NN, H, H, 0);
    ln_kernel<<<NN / 4, 128, 0, stream>>>(tmpF, dg + ((size_t)l * 2 + 0) * H,
                                          db + ((size_t)l * 2 + 0) * H, nullptr,
                                          hVf, hVb, NN);
    gemmb(hVb, dInT[l], dbin + (size_t)l * FF, nullptr, nullptr, midB, NN, FF, H, 1);
    gemmb(midB, dOutT[l], dbout + (size_t)l * H, hVf, tmpF, nullptr, NN, H, FF, 0);
    ln_kernel<<<NN / 4, 128, 0, stream>>>(tmpF, dg + ((size_t)l * 2 + 1) * H,
                                          db + ((size_t)l * 2 + 1) * H, mask,
                                          hVf, hVb, NN);
  }

  // ---- output head (N=20 padded to 2 col tiles) ----
  {
    const long waves = (long)(NN / 16) * 1;
    gemm_kernel<bf16, 2><<<dim3((unsigned)((waves + 3) / 4)), 128, 0, stream>>>(
        hVb, WoT, b_out, nullptr, logF, nullptr, NN, VOC, H, 0);
  }
  lsm_kernel<<<(NN + 127) / 128, 128, 0, stream>>>(logF, (float*)d_out, NN, VOC);
}
